// MultiHeadAttention_36163624632535
// MI455X (gfx1250) — matrix operand order repaired and checked by reading, not compiled, not run on hardware
//
#include <hip/hip_runtime.h>
#include <hip/hip_bf16.h>

typedef __attribute__((ext_vector_type(16))) __bf16 v16bf;
typedef __attribute__((ext_vector_type(8)))  __bf16 v8bf;
typedef __attribute__((ext_vector_type(8)))  float  v8f;

#define WMMA_BF16(a, b, c) \
  __builtin_amdgcn_wmma_f32_16x16x32_bf16(false, (a), false, (b), (short)0, (c), false, false)

static constexpr int BSZ  = 4;
static constexpr int NSEQ = 2048;
static constexpr int EMB  = 1024;
static constexpr int HNUM = 16;
static constexpr int DH   = 64;
static constexpr int ROWS = BSZ * NSEQ;      // 8192
static constexpr float LOG2E = 1.44269504088896f;

// ---------------------------------------------------------------------------
// A-fragment loader: 16x32 bf16 tile, row-major source, leading elements at
// rowptr (= base + row*lda + k0). In this code's contiguous k order, lane element i has
// k = 16*hi + i  -> two contiguous 8-element (16B) loads.
// ---------------------------------------------------------------------------
__device__ inline v16bf load_afrag(const __bf16* rowptr, int hi) {
  v8bf lo = *(const v8bf*)(rowptr + 16 * hi);
  v8bf hh = *(const v8bf*)(rowptr + 16 * hi + 8);
  return __builtin_shufflevector(lo, hh, 0, 1, 2, 3, 4, 5, 6, 7,
                                 8, 9, 10, 11, 12, 13, 14, 15);
}

// ---------------------------------------------------------------------------
// fp32 -> bf16 elementwise convert
// ---------------------------------------------------------------------------
__global__ __launch_bounds__(256) void cvt_kernel(const float* __restrict__ s,
                                                  __bf16* __restrict__ d, int n) {
  int i = blockIdx.x * blockDim.x + threadIdx.x;
  if (i < n) d[i] = (__bf16)s[i];
}

// fp32 [1024,1024] -> bf16 transposed: d[n*1024 + k] = s[k*1024 + n]
__global__ __launch_bounds__(256) void cvtT_kernel(const float* __restrict__ s,
                                                   __bf16* __restrict__ d) {
  int i = blockIdx.x * blockDim.x + threadIdx.x;
  int k = i >> 10;
  int n = i & (EMB - 1);
  d[(size_t)n * EMB + k] = (__bf16)s[i];
}

// ---------------------------------------------------------------------------
// QKV projection GEMM: C[8192,1024] = xb[8192,1024] * W (W given transposed).
// blockIdx.z selects Wq/Wk/Wv. 256 threads = 8 waves, 128x128 tile per block,
// each wave computes 32x64 via 2x4 grid of 16x16 WMMA accumulators.
// Q is pre-scaled by 1/sqrt(64); V is stored d-major for the P*V GEMM.
// ---------------------------------------------------------------------------
__global__ __launch_bounds__(256)
void gemm_qkv(const __bf16* __restrict__ A, const __bf16* __restrict__ WT,
              __bf16* __restrict__ Qh, __bf16* __restrict__ Kh,
              __bf16* __restrict__ Vt) {
  const int z    = blockIdx.z;
  const __bf16* W = WT + (size_t)z * EMB * EMB;
  const int lane = threadIdx.x & 31;
  const int wave = threadIdx.x >> 5;
  const int m15  = lane & 15;
  const int hi   = lane >> 4;
  const int BM   = blockIdx.x * 128 + (wave >> 1) * 32;
  const int BN   = blockIdx.y * 128 + (wave & 1) * 64;

  v8f acc[2][4] = {};

  for (int k0 = 0; k0 < EMB; k0 += 32) {
    v16bf a0 = load_afrag(A + (size_t)(BM + m15) * EMB + k0, hi);
    v16bf a1 = load_afrag(A + (size_t)(BM + 16 + m15) * EMB + k0, hi);
#pragma unroll
    for (int nt = 0; nt < 4; ++nt) {
      v16bf b = *(const v16bf*)(W + (size_t)(BN + nt * 16 + m15) * EMB + k0 + 16 * hi);
      acc[0][nt] = WMMA_BF16(a0, b, acc[0][nt]);
      acc[1][nt] = WMMA_BF16(a1, b, acc[1][nt]);
    }
  }

  const float scale = (z == 0) ? 0.125f : 1.0f;  // fold 1/sqrt(64) into Q
#pragma unroll
  for (int mt = 0; mt < 2; ++mt) {
#pragma unroll
    for (int nt = 0; nt < 4; ++nt) {
#pragma unroll
      for (int r = 0; r < 8; ++r) {
        int g = BM + mt * 16 + r + 8 * hi;       // global row in [0, 8192)
        int c = BN + nt * 16 + m15;              // global col in [0, 1024)
        int b = g >> 11, n = g & (NSEQ - 1);
        int h = c >> 6,  d = c & (DH - 1);
        size_t bh = (size_t)b * HNUM + h;
        __bf16 v = (__bf16)(acc[mt][nt][r] * scale);
        if (z == 0)      Qh[(bh * NSEQ + n) * DH + d] = v;
        else if (z == 1) Kh[(bh * NSEQ + n) * DH + d] = v;
        else             Vt[(bh * DH + d) * NSEQ + n] = v;
      }
    }
  }
}

// ---------------------------------------------------------------------------
// Flash attention. Block = 128 threads = 4 waves; each wave owns 16 query
// rows; block covers 64 queries of one (b,h). 32-key steps:
//   S(16x32) = Q(16x64) * K^T  (4 WMMA),  online softmax in fp32,
//   P staged via LDS (C-layout -> A-layout), O += P * V (4 WMMA).
// ---------------------------------------------------------------------------
__global__ __launch_bounds__(128)
void attn_kernel(const __bf16* __restrict__ Qh, const __bf16* __restrict__ Kh,
                 const __bf16* __restrict__ Vt, __bf16* __restrict__ AO) {
  __shared__ __align__(64) __bf16 plds[4][16][32];

  const int lane = threadIdx.x & 31;
  const int wave = threadIdx.x >> 5;
  const int m15  = lane & 15;
  const int hi   = lane >> 4;
  const int bh   = blockIdx.y;                 // 0..63
  const int q0   = blockIdx.x * 64 + wave * 16;

  const __bf16* Qb = Qh + (size_t)bh * NSEQ * DH;
  const __bf16* Kb = Kh + (size_t)bh * NSEQ * DH;
  const __bf16* Vb = Vt + (size_t)bh * DH * NSEQ;

  const __bf16* qrow = Qb + (size_t)(q0 + m15) * DH;
  const v16bf aq0 = load_afrag(qrow, hi);        // d = 0..31
  const v16bf aq1 = load_afrag(qrow + 32, hi);   // d = 32..63

  v8f o0 = {}, o1 = {}, o2 = {}, o3 = {};
  float m_r[8], l_r[8];
#pragma unroll
  for (int r = 0; r < 8; ++r) { m_r[r] = -3.0e38f; l_r[r] = 0.0f; }

  for (int key0 = 0; key0 < NSEQ; key0 += 32) {
    // ---- S = Q * K^T (scale already folded into Q) ----
    v8f s0 = {}, s1 = {};
    {
      const __bf16* kp0 = Kb + (size_t)(key0 + m15) * DH + 16 * hi;
      const __bf16* kp1 = Kb + (size_t)(key0 + 16 + m15) * DH + 16 * hi;
      v16bf bk;
      bk = *(const v16bf*)(kp0);      s0 = WMMA_BF16(aq0, bk, s0);
      bk = *(const v16bf*)(kp0 + 32); s0 = WMMA_BF16(aq1, bk, s0);
      bk = *(const v16bf*)(kp1);      s1 = WMMA_BF16(aq0, bk, s1);
      bk = *(const v16bf*)(kp1 + 32); s1 = WMMA_BF16(aq1, bk, s1);
    }

    // ---- online softmax over this 32-key panel ----
#pragma unroll
    for (int r = 0; r < 8; ++r) {
      float rm = fmaxf(s0[r], s1[r]);
      rm = fmaxf(rm, __shfl_xor(rm, 1));
      rm = fmaxf(rm, __shfl_xor(rm, 2));
      rm = fmaxf(rm, __shfl_xor(rm, 4));
      rm = fmaxf(rm, __shfl_xor(rm, 8));
      float mnew = fmaxf(m_r[r], rm);
      float corr = exp2f((m_r[r] - mnew) * LOG2E);
      o0[r] *= corr; o1[r] *= corr; o2[r] *= corr; o3[r] *= corr;
      float p0 = exp2f((s0[r] - mnew) * LOG2E);
      float p1 = exp2f((s1[r] - mnew) * LOG2E);
      plds[wave][r + 8 * hi][m15]      = (__bf16)p0;
      plds[wave][r + 8 * hi][m15 + 16] = (__bf16)p1;
      float ps = p0 + p1;
      ps += __shfl_xor(ps, 1);
      ps += __shfl_xor(ps, 2);
      ps += __shfl_xor(ps, 4);
      ps += __shfl_xor(ps, 8);
      l_r[r] = l_r[r] * corr + ps;
      m_r[r] = mnew;
    }

    // P: LDS C-layout staging -> A-layout fragment (per-wave private region;
    // DS ops are in-order per wave, wait out the stores explicitly).
    asm volatile("s_wait_dscnt 0" ::: "memory");
    const __bf16* prow = &plds[wave][m15][0];
    v8bf plo = *(const v8bf*)(prow + 16 * hi);
    v8bf phi = *(const v8bf*)(prow + 16 * hi + 8);
    v16bf ap = __builtin_shufflevector(plo, phi, 0, 1, 2, 3, 4, 5, 6, 7,
                                       8, 9, 10, 11, 12, 13, 14, 15);

    // ---- O += P * V  (V stored d-major -> contiguous B fragments) ----
    const __bf16* vp = Vb + key0 + 16 * hi;
    o0 = WMMA_BF16(ap, *(const v16bf*)(vp + (size_t)(m15) * NSEQ),      o0);
    o1 = WMMA_BF16(ap, *(const v16bf*)(vp + (size_t)(16 + m15) * NSEQ), o1);
    o2 = WMMA_BF16(ap, *(const v16bf*)(vp + (size_t)(32 + m15) * NSEQ), o2);
    o3 = WMMA_BF16(ap, *(const v16bf*)(vp + (size_t)(48 + m15) * NSEQ), o3);
  }

  // ---- epilogue: O / l, concat heads back to [B*N, E] (bf16) ----
  const int b = bh >> 4, h = bh & 15;
#pragma unroll
  for (int r = 0; r < 8; ++r) {
    float inv = 1.0f / l_r[r];
    int q = q0 + r + 8 * hi;
    size_t off = ((size_t)(b * NSEQ + q)) * EMB + h * DH;
    AO[off + m15]      = (__bf16)(o0[r] * inv);
    AO[off + 16 + m15] = (__bf16)(o1[r] * inv);
    AO[off + 32 + m15] = (__bf16)(o2[r] * inv);
    AO[off + 48 + m15] = (__bf16)(o3[r] * inv);
  }
}

// ---------------------------------------------------------------------------
// Output projection: out[8192,1024] = AO * Wo + bo  (fp32 out)
// ---------------------------------------------------------------------------
__global__ __launch_bounds__(256)
void gemm_out(const __bf16* __restrict__ A, const __bf16* __restrict__ WT,
              const float* __restrict__ bias, float* __restrict__ out) {
  const int lane = threadIdx.x & 31;
  const int wave = threadIdx.x >> 5;
  const int m15  = lane & 15;
  const int hi   = lane >> 4;
  const int BM   = blockIdx.x * 128 + (wave >> 1) * 32;
  const int BN   = blockIdx.y * 128 + (wave & 1) * 64;

  v8f acc[2][4] = {};

  for (int k0 = 0; k0 < EMB; k0 += 32) {
    v16bf a0 = load_afrag(A + (size_t)(BM + m15) * EMB + k0, hi);
    v16bf a1 = load_afrag(A + (size_t)(BM + 16 + m15) * EMB + k0, hi);
#pragma unroll
    for (int nt = 0; nt < 4; ++nt) {
      v16bf b = *(const v16bf*)(WT + (size_t)(BN + nt * 16 + m15) * EMB + k0 + 16 * hi);
      acc[0][nt] = WMMA_BF16(a0, b, acc[0][nt]);
      acc[1][nt] = WMMA_BF16(a1, b, acc[1][nt]);
    }
  }

#pragma unroll
  for (int mt = 0; mt < 2; ++mt) {
#pragma unroll
    for (int nt = 0; nt < 4; ++nt) {
#pragma unroll
      for (int r = 0; r < 8; ++r) {
        int g = BM + mt * 16 + r + 8 * hi;
        int c = BN + nt * 16 + m15;
        out[(size_t)g * EMB + c] = acc[mt][nt][r] + bias[c];
      }
    }
  }
}

// ---------------------------------------------------------------------------
extern "C" void kernel_launch(void* const* d_in, const int* in_sizes, int n_in,
                              void* d_out, int out_size, void* d_ws, size_t ws_size,
                              hipStream_t stream) {
  const float* x  = (const float*)d_in[0];
  const float* Wq = (const float*)d_in[1];
  const float* Wk = (const float*)d_in[2];
  const float* Wv = (const float*)d_in[3];
  const float* Wo = (const float*)d_in[4];
  const float* bo = (const float*)d_in[5];
  float* out = (float*)d_out;

  // Workspace layout (bf16), total ~92 MB
  __bf16* xb  = (__bf16*)d_ws;                            // [8192,1024]
  __bf16* wT  = xb  + (size_t)ROWS * EMB;                 // 3x [1024,1024] (transposed Wq,Wk,Wv)
  __bf16* woT = wT  + (size_t)3 * EMB * EMB;              // [1024,1024] transposed Wo
  __bf16* qh  = woT + (size_t)EMB * EMB;                  // [64, 2048, 64]
  __bf16* kh  = qh  + (size_t)64 * NSEQ * DH;             // [64, 2048, 64]
  __bf16* vt  = kh  + (size_t)64 * NSEQ * DH;             // [64, 64, 2048] (d-major)
  __bf16* ao  = vt  + (size_t)64 * NSEQ * DH;             // [8192, 1024]

  const int nX = ROWS * EMB;      // 8388608
  const int nW = EMB * EMB;       // 1048576
  cvt_kernel <<<(nX + 255) / 256, 256, 0, stream>>>(x, xb, nX);
  cvtT_kernel<<<(nW + 255) / 256, 256, 0, stream>>>(Wq, wT + 0 * (size_t)nW);
  cvtT_kernel<<<(nW + 255) / 256, 256, 0, stream>>>(Wk, wT + 1 * (size_t)nW);
  cvtT_kernel<<<(nW + 255) / 256, 256, 0, stream>>>(Wv, wT + 2 * (size_t)nW);
  cvtT_kernel<<<(nW + 255) / 256, 256, 0, stream>>>(Wo, woT);

  gemm_qkv<<<dim3(ROWS / 128, EMB / 128, 3), 256, 0, stream>>>(xb, wT, qh, kh, vt);
  attn_kernel<<<dim3(NSEQ / 64, BSZ * HNUM), 128, 0, stream>>>(qh, kh, vt, ao);
  gemm_out<<<dim3(ROWS / 128, EMB / 128), 256, 0, stream>>>(ao, woT, bo, out);
}
